// NuggetScorer_54400055771153
// MI455X (gfx1250) — matrix operand order repaired and checked by reading, not compiled, not run on hardware
//
#include <hip/hip_runtime.h>
#include <math.h>

// ---------------------------------------------------------------------------
// NuggetScorer for MI455X (gfx1250, wave32, WMMA)
//
//   B=8, S=4096, D=2048, H=128, M = B*S = 32768, max_nugget = 410
//
// Phase 1: scores = (GELU(X @ w1 + b1)) @ w2 + b2
//   17.2 GFLOP GEMM, HBM floor = 256 MB read of X (~11 us @ 23.3 TB/s).
//   v_wmma_f32_16x16x32_f16 with on-the-fly f32->f16 conversion, LDS
//   double-buffered + register prefetch so the HBM stream never stalls
//   behind the WMMA pipeline (one barrier per K-chunk).
// Phase 2: per-batch bitonic argsort (desc score, asc index) of 4096 keys.
// Phase 3: gather of top-410 rows of hidden_states (bandwidth only).
// ---------------------------------------------------------------------------

typedef __attribute__((ext_vector_type(16))) _Float16 v16h;
typedef __attribute__((ext_vector_type(8)))  float    v8f;

#define B_   8
#define S_   4096
#define D_   2048
#define H_   128
#define M_   (B_ * S_)          // 32768 rows in the GEMM
#define MAXN 410                // ceil(S * 0.1)
#define NK   (D_ / 32)          // 64 K-chunks of 32

// d_out = outputs concatenated flat (all float32):
//   enc (B,410,2048) | nugget_mask (B,410) | nugget_scores (B,410)
//   | indices (B,410) | scores (B,S)
#define OFF_ENC  0
#define OFF_MASK (B_ * MAXN * D_)            // 6,717,440
#define OFF_NSC  (OFF_MASK + B_ * MAXN)      // 6,720,720
#define OFF_IDX  (OFF_NSC  + B_ * MAXN)      // 6,724,000
#define OFF_SC   (OFF_IDX  + B_ * MAXN)      // 6,727,280  (+32768 = 6,760,048)

// ---------------------------------------------------------------------------
// Kernel 1: fused score MLP.
// Block = 256 threads (8 wave32). Each block owns a 128-row M-tile, N=128.
// ---------------------------------------------------------------------------
__global__ __launch_bounds__(256, 2)
void nugget_score_gemm(const float* __restrict__ lhs,   // (M, D) row-major
                       const float* __restrict__ w1,    // (D, H) row-major
                       const float* __restrict__ b1,    // (H)
                       const float* __restrict__ w2,    // (H)
                       const float* __restrict__ b2,    // (1)
                       float* __restrict__ out_scores)  // (M)
{
    __shared__ _Float16 As[2 * 128 * 32];   // A tiles, f16 (2 x 8 KB)
    __shared__ _Float16 Ws[2 * 32 * 128];   // W tiles, f16 (2 x 8 KB)
    __shared__ float    red[128 * 16];      // epilogue partials (8 KB)
    __shared__ float    b1s[H_], w2s[H_];

    const int tid  = threadIdx.x;
    const int wave = tid >> 5;          // 0..7  -> 16-row slice of the M-tile
    const int lane = tid & 31;
    const int m0   = blockIdx.x * 128;  // global row base

    if (tid < H_) { b1s[tid] = b1[tid]; w2s[tid] = w2[tid]; }

    // --- loop-invariant staging coordinates (4 float4 each for A and W) ---
    // A tile: 128 rows x 32 K  = 1024 float4 ; thread e = tid + i*256
    // W tile:  32 rows x 128 H = 1024 float4
    const float* gA[4];
    const float* gW[4];
    _Float16*    sA[4];
    _Float16*    sW[4];
#pragma unroll
    for (int i = 0; i < 4; ++i) {
        int e = tid + i * 256;
        int ar = e >> 3, aq = e & 7;       // A: 8 float4 per row
        int wr = e >> 5, wq = e & 31;      // W: 32 float4 per row
        gA[i] = lhs + (size_t)(m0 + ar) * D_ + aq * 4;
        gW[i] = w1 + (size_t)wr * H_ + wq * 4;
        sA[i] = &As[ar * 32 + aq * 4];
        sW[i] = &Ws[wr * H_ + wq * 4];
    }

    v8f acc[8];
    {
        v8f z = {};
#pragma unroll
        for (int n = 0; n < 8; ++n) acc[n] = z;
    }

    // WMMA f16 A-fragment addressing (ISA 7.12.2, 16-bit A 16x32):
    //   lanes 0-15 : M = lane,    VGPR0-3 = K 0..7,  VGPR4-7 = K 16..23
    //   lanes 16-31: M = lane-16, VGPR0-3 = K 8..15, VGPR4-7 = K 24..31
    const int aoff = (wave * 16 + (lane & 15)) * 32 + (lane >> 4) * 8;
    const int boff = (lane & 15) + (lane >> 4) * 8 * H_;

    float4 ra[4], rw[4];

    // prologue: fetch chunk 0 and stage into buffer 0
#pragma unroll
    for (int i = 0; i < 4; ++i) {
        ra[i] = *(const float4*)(gA[i]);
        rw[i] = *(const float4*)(gW[i]);
    }
#pragma unroll
    for (int i = 0; i < 4; ++i) {
        union { _Float16 h[4]; int2 d; } pa, pw;
        pa.h[0] = (_Float16)ra[i].x; pa.h[1] = (_Float16)ra[i].y;
        pa.h[2] = (_Float16)ra[i].z; pa.h[3] = (_Float16)ra[i].w;
        pw.h[0] = (_Float16)rw[i].x; pw.h[1] = (_Float16)rw[i].y;
        pw.h[2] = (_Float16)rw[i].z; pw.h[3] = (_Float16)rw[i].w;
        *(int2*)(sA[i]) = pa.d;
        *(int2*)(sW[i]) = pw.d;
    }
    __syncthreads();

    for (int kc = 0; kc < NK; ++kc) {
        const int cur = kc & 1;

        // --- prefetch chunk kc+1 into registers (HBM latency hidden by WMMA)
        if (kc + 1 < NK) {
            const int k1 = (kc + 1) * 32;
#pragma unroll
            for (int i = 0; i < 4; ++i) {
                ra[i] = *(const float4*)(gA[i] + k1);
                rw[i] = *(const float4*)(gW[i] + (size_t)k1 * H_);
                __builtin_prefetch(gA[i] + k1 + 32, 0, 1);  // warm chunk kc+2
            }
        }

        // --- compute from buffer `cur`
        const _Float16* Ab = &As[cur * (128 * 32)];
        const _Float16* Wb = &Ws[cur * (32 * 128)];

        union { v16h v; int4 q[2]; } af;
        af.q[0] = *(const int4*)(Ab + aoff);
        af.q[1] = *(const int4*)(Ab + aoff + 16);

#pragma unroll
        for (int n = 0; n < 8; ++n) {
            // B 32x16: lane%16 = N col; elems K 8h..8h+7, 16+8h..23+8h
            union { v16h v; int4 q[2]; } bf;
            const _Float16* Bp = Wb + boff + n * 16;
#pragma unroll
            for (int s = 0; s < 16; ++s)
                bf.v[s] = Bp[(s < 8 ? s : s + 8) * H_];
            acc[n] = __builtin_amdgcn_wmma_f32_16x16x32_f16(
                false, af.v, false, bf.v, (short)0, acc[n], false, false);
        }

        // --- stage chunk kc+1 into the alternate buffer; one barrier/iter.
        // WAR on buffer cur^1 is protected by the previous iteration's barrier.
        if (kc + 1 < NK) {
            const int off = (cur ^ 1) * (128 * 32);   // same elem count for W
#pragma unroll
            for (int i = 0; i < 4; ++i) {
                union { _Float16 h[4]; int2 d; } pa, pw;
                pa.h[0] = (_Float16)ra[i].x; pa.h[1] = (_Float16)ra[i].y;
                pa.h[2] = (_Float16)ra[i].z; pa.h[3] = (_Float16)ra[i].w;
                pw.h[0] = (_Float16)rw[i].x; pw.h[1] = (_Float16)rw[i].y;
                pw.h[2] = (_Float16)rw[i].z; pw.h[3] = (_Float16)rw[i].w;
                *(int2*)(sA[i] + off) = pa.d;
                *(int2*)(sW[i] + off) = pw.d;
            }
            __syncthreads();
        }
    }

    // --- epilogue: bias + exact GELU + dot with w2, fused in registers.
    // C/D layout: VGPR v, lanes 0-15 -> (M=v, N=lane); lanes 16-31 -> (M=v+8).
    const int col = lane & 15;
    const int rb  = (lane >> 4) * 8;
#pragma unroll
    for (int v = 0; v < 8; ++v) {
        float partial = 0.f;
#pragma unroll
        for (int n = 0; n < 8; ++n) {
            float x = acc[n][v] + b1s[n * 16 + col];
            float g = 0.5f * x * (1.f + erff(x * 0.70710678118654752f));
            partial += g * w2s[n * 16 + col];
        }
        red[(wave * 16 + rb + v) * 16 + col] = partial;
    }
    __syncthreads();
    if (tid < 128) {
        float s = b2[0];
#pragma unroll
        for (int c = 0; c < 16; ++c) s += red[tid * 16 + c];
        out_scores[m0 + tid] = s;
    }
}

// ---------------------------------------------------------------------------
// Kernel 2: per-batch descending stable argsort via bitonic sort of u64 keys
//   key = (~orderable(score) << 32) | index  -> ascending sort gives
//   descending score with ascending-index tie-break (matches stable argsort).
// Also computes n_token -> n_nugget -> nugget_mask.
// ---------------------------------------------------------------------------
__global__ __launch_bounds__(1024, 1)
void nugget_topk(const int* __restrict__ mask, float* __restrict__ out)
{
    __shared__ unsigned long long keys[S_];   // 32 KB
    __shared__ int ntok;

    const int b   = blockIdx.x;
    const int tid = threadIdx.x;
    const float* sc = out + OFF_SC + (size_t)b * S_;

    if (tid == 0) ntok = 0;
    __syncthreads();

    int local = 0;
    for (int i = tid; i < S_; i += 1024) {
        unsigned u = __float_as_uint(sc[i]);
        u = (u & 0x80000000u) ? ~u : (u | 0x80000000u);   // order-preserving map
        keys[i] = ((unsigned long long)(~u) << 32) | (unsigned)i;
        local += mask[b * S_ + i];
    }
    atomicAdd(&ntok, local);

    for (int k = 2; k <= S_; k <<= 1) {
        for (int j = k >> 1; j > 0; j >>= 1) {
            __syncthreads();
            for (int i = tid; i < S_; i += 1024) {
                int ixj = i ^ j;
                if (ixj > i) {
                    unsigned long long a = keys[i], c = keys[ixj];
                    bool up = ((i & k) == 0);
                    if ((a > c) == up) { keys[i] = c; keys[ixj] = a; }
                }
            }
        }
    }
    __syncthreads();

    int nt = ntok;
    int nn = (int)ceilf((float)nt * 0.1f);
    nn = nn < 1 ? 1 : nn;
    nn = nn > nt ? nt : nn;

    for (int j = tid; j < MAXN; j += 1024) {
        unsigned idx = (unsigned)(keys[j] & 0xFFFFFFFFu);
        out[OFF_NSC  + b * MAXN + j] = sc[idx];
        out[OFF_IDX  + b * MAXN + j] = (float)idx;
        out[OFF_MASK + b * MAXN + j] = (j < nn) ? 1.0f : 0.0f;
    }
}

// ---------------------------------------------------------------------------
// Kernel 3: enc gather. One block per (b, j): copy hidden_states[b, idx, :].
// Index is read back from the float `indices` output (exact for idx < 2^24).
// ---------------------------------------------------------------------------
__global__ __launch_bounds__(256)
void nugget_gather(const float* __restrict__ hid, float* __restrict__ out)
{
    const int bj  = blockIdx.x;           // 0 .. B*MAXN-1
    const int b   = bj / MAXN;
    const int tid = threadIdx.x;
    const int idx = (int)out[OFF_IDX + bj];

    const float4* src = (const float4*)(hid + ((size_t)b * S_ + idx) * D_);
    float4*       dst = (float4*)(out + OFF_ENC + (size_t)bj * D_);
#pragma unroll
    for (int i = 0; i < 2; ++i)           // 512 float4 per row, 256 threads
        dst[tid + i * 256] = src[tid + i * 256];
}

// ---------------------------------------------------------------------------
extern "C" void kernel_launch(void* const* d_in, const int* in_sizes, int n_in,
                              void* d_out, int out_size, void* d_ws, size_t ws_size,
                              hipStream_t stream)
{
    const float* lhs  = (const float*)d_in[0];   // last_hidden_state (B,S,D)
    const float* hid  = (const float*)d_in[1];   // hidden_states     (B,S,D)
    const int*   msk  = (const int*)  d_in[2];   // attention_mask    (B,S)
    const float* w1   = (const float*)d_in[3];   // (D,H)
    const float* b1   = (const float*)d_in[4];   // (H)
    const float* w2   = (const float*)d_in[5];   // (H,1)
    const float* b2   = (const float*)d_in[6];   // (1)
    float* out = (float*)d_out;

    nugget_score_gemm<<<M_ / 128, 256, 0, stream>>>(lhs, w1, b1, w2, b2,
                                                    out + OFF_SC);
    nugget_topk<<<B_, 1024, 0, stream>>>(msk, out);
    nugget_gather<<<B_ * MAXN, 256, 0, stream>>>(hid, out);
}
